// LSTMAttention_4191888081444
// MI455X (gfx1250) — matrix operand order repaired and checked by reading, not compiled, not run on hardware
//
#include <hip/hip_runtime.h>
#include <math.h>

#define B_ 64
#define S_ 256
#define E_ 512
#define H_ 512
#define C_ 20
#define G4H 2048   // 4*H

typedef __attribute__((ext_vector_type(16))) __bf16        v16bf;
typedef __attribute__((ext_vector_type(8)))  float         v8f;
typedef __attribute__((ext_vector_type(8)))  unsigned int  v8u;
typedef __attribute__((ext_vector_type(4)))  unsigned int  v4u;

// ---------- helpers ----------
__device__ __forceinline__ unsigned short f2bf(float f) {
    unsigned int u = __builtin_bit_cast(unsigned int, f);
    unsigned int r = u + 0x7FFFu + ((u >> 16) & 1u);   // round-to-nearest-even
    return (unsigned short)(r >> 16);
}

__device__ __forceinline__ float sigm(float x) { return 1.0f / (1.0f + __expf(-x)); }

// A-fragment (16x32 bf16): lane holds row m, K = kh+{0..7} and 16+kh+{0..7}, kh = (lane<16?0:8)
// p already points at row + k0 + kh (16B aligned); works for global or LDS pointers
__device__ __forceinline__ v16bf ld_afrag(const unsigned short* __restrict__ p) {
    v4u lo = *(const v4u*)p;          // K kh+0..7   -> VGPR 0..3
    v4u hi = *(const v4u*)(p + 16);   // K 16+kh+0..7 -> VGPR 4..7
    v8u d;
    d[0]=lo[0]; d[1]=lo[1]; d[2]=lo[2]; d[3]=lo[3];
    d[4]=hi[0]; d[5]=hi[1]; d[6]=hi[2]; d[7]=hi[3];
    return __builtin_bit_cast(v16bf, d);
}

// B-fragment (32x16 bf16): lane holds col n, K = kh+{0..7} and 16+kh+{0..7}, kh = (lane<16?0:8)
__device__ __forceinline__ v16bf ld_bfrag(const unsigned short* __restrict__ p) {
    v4u lo = *(const v4u*)p;
    v4u hi = *(const v4u*)(p + 16);
    v8u d;
    d[0]=lo[0]; d[1]=lo[1]; d[2]=lo[2]; d[3]=lo[3];
    d[4]=hi[0]; d[5]=hi[1]; d[6]=hi[2]; d[7]=hi[3];
    return __builtin_bit_cast(v16bf, d);
}

#define WMMA_BF16(a, b, c) \
    __builtin_amdgcn_wmma_f32_16x16x32_bf16(false, (a), false, (b), (short)0, (c), false, false)

// ---------- conversion / gather / init ----------
__global__ void k_f2bf(const float* __restrict__ src, unsigned short* __restrict__ dst, int n) {
    int i = blockIdx.x * 256 + threadIdx.x;
    if (i < n) dst[i] = f2bf(src[i]);
}

__global__ void k_embed(const int* __restrict__ x, const float* __restrict__ embed,
                        unsigned short* __restrict__ out) {
    long long i = (long long)blockIdx.x * 256 + threadIdx.x;   // B*S*E total
    if (i >= (long long)B_ * S_ * E_) return;
    int e = (int)(i % E_);
    long long bs = i / E_;
    int tok = x[bs];
    out[i] = f2bf(embed[(long long)tok * E_ + e]);
}

__global__ void k_init(unsigned short* __restrict__ hs, float* __restrict__ cs,
                       unsigned int* __restrict__ sync) {
    int i = blockIdx.x * 256 + threadIdx.x;
    if (i < 4 * B_ * H_) hs[i] = 0;        // 2 dirs x 2 buffers
    if (i < 2 * B_ * H_) cs[i] = 0.0f;
    if (i < 2) sync[i] = 0u;
}

// ---------- input-projection GEMM: xproj[dir] = emb @ w_ih^T + b_ih ----------
// grid (8, 256, 2), block 256. WG computes 64 rows x 256 cols; wave w -> cols [w*32, w*32+32)
__global__ void __launch_bounds__(256)
k_xproj(const unsigned short* __restrict__ embB,
        const unsigned short* __restrict__ wih_f, const unsigned short* __restrict__ wih_b,
        const float* __restrict__ bih_f, const float* __restrict__ bih_b,
        float* __restrict__ xp_f, float* __restrict__ xp_b) {
    const int dir = blockIdx.z;
    const unsigned short* W = dir ? wih_b : wih_f;
    const float* bias       = dir ? bih_b : bih_f;
    float* C                = dir ? xp_b  : xp_f;

    const int row0 = blockIdx.y * 64;
    const int col0 = blockIdx.x * 256;
    const int tid  = threadIdx.x;
    const int wave = tid >> 5, lane = tid & 31;
    const int mrow = lane & 15, hi = lane >> 4;
    const int khA = hi * 8, khB = hi * 8;

    int n[2];
    n[0] = col0 + wave * 32 + (lane & 15);
    n[1] = n[0] + 16;

    v8f acc[4][2];
    for (int ct = 0; ct < 2; ++ct) {
        float bv = bias[n[ct]];
        for (int rt = 0; rt < 4; ++rt)
            for (int r = 0; r < 8; ++r) acc[rt][ct][r] = bv;
    }

    const unsigned short* Arow[4];
    for (int rt = 0; rt < 4; ++rt)
        Arow[rt] = embB + (size_t)(row0 + rt * 16 + mrow) * E_;
    const unsigned short* Brow[2];
    for (int ct = 0; ct < 2; ++ct)
        Brow[ct] = W + (size_t)n[ct] * E_;

    for (int k0 = 0; k0 < E_; k0 += 32) {
        v16bf bf[2], af[4];
        for (int ct = 0; ct < 2; ++ct) bf[ct] = ld_bfrag(Brow[ct] + k0 + khB);
        for (int rt = 0; rt < 4; ++rt) af[rt] = ld_afrag(Arow[rt] + k0 + khA);
        for (int rt = 0; rt < 4; ++rt)
            for (int ct = 0; ct < 2; ++ct)
                acc[rt][ct] = WMMA_BF16(af[rt], bf[ct], acc[rt][ct]);
    }

    for (int rt = 0; rt < 4; ++rt)
        for (int ct = 0; ct < 2; ++ct)
            for (int r = 0; r < 8; ++r) {
                int m = row0 + rt * 16 + r + 8 * hi;
                C[(size_t)m * G4H + n[ct]] = acc[rt][ct][r];
            }
}

// ---------- persistent bidirectional LSTM recurrence ----------
// grid (8, 2), block 256. 8 WGs per direction, each owns 64 cols of H across all 4 gates.
// h_state double-buffered -> ONE device-scope barrier per step.
__device__ __forceinline__ void dir_sync(unsigned int* cnt, unsigned int ep) {
    __syncthreads();
    __threadfence();                               // release prior global writes
    if (threadIdx.x == 0) {
        atomicAdd(cnt, 1u);
        while (*(volatile unsigned int*)cnt < 8u * ep)
            __builtin_amdgcn_s_sleep(1);
    }
    __syncthreads();
    __threadfence();                               // acquire remote writes
}

__global__ void __launch_bounds__(256)
k_recurrent(const float* __restrict__ xp_f, const float* __restrict__ xp_b,
            const unsigned short* __restrict__ whh_f, const unsigned short* __restrict__ whh_b,
            const float* __restrict__ bhh_f, const float* __restrict__ bhh_b,
            unsigned short* __restrict__ h_state, float* __restrict__ c_state,
            float* __restrict__ hseq_f, float* __restrict__ hseq_b,
            unsigned int* __restrict__ sync) {
    const int dir = blockIdx.y;
    const int cb  = blockIdx.x;                    // 64-col block of H
    const float* xp          = dir ? xp_b  : xp_f;
    const unsigned short* W  = dir ? whh_b : whh_f;
    const float* bhh         = dir ? bhh_b : bhh_f;
    float* hseq              = dir ? hseq_b : hseq_f;
    unsigned short* hs0 = h_state + ((size_t)dir * 2 + 0) * B_ * H_;
    unsigned short* hs1 = h_state + ((size_t)dir * 2 + 1) * B_ * H_;
    float* cs           = c_state + (size_t)dir * B_ * H_;
    unsigned int* cnt   = sync + dir;

    const int tid  = threadIdx.x;
    const int wave = tid >> 5, lane = tid & 31;
    const int mrow = lane & 15, hi = lane >> 4;
    const int gate = wave >> 1;
    const int khA = hi * 8, khB = hi * 8;

    int lc[2], n[2];
    lc[0] = wave * 32 + (lane & 15);  lc[1] = lc[0] + 16;                 // local col 0..255
    n[0]  = gate * H_ + cb * 64 + (wave & 1) * 32 + (lane & 15);          // global gate col
    n[1]  = n[0] + 16;

    const unsigned short* Brow[2];
    for (int ct = 0; ct < 2; ++ct) Brow[ct] = W + (size_t)n[ct] * H_;

    // 64 KB LDS, reused per step: phase 1 = h(t-1) bf16 [64][512]; phase 2 = g f32 [64][256]
    __shared__ v4u lds_buf[4096];
    unsigned short* h_lds = (unsigned short*)lds_buf;
    float*          g_lds = (float*)lds_buf;
    const unsigned  lds_base = (unsigned)(size_t)(void*)h_lds;   // low 32 bits = LDS offset

    for (int t = 0; t < S_; ++t) {
        const int s = dir ? (S_ - 1 - t) : t;
        const unsigned short* hs_rd = (t & 1) ? hs1 : hs0;
        unsigned short*       hs_wr = (t & 1) ? hs0 : hs1;

        // ---- phase 1: async-stage h(t-1) (64x512 bf16 = 64 KB) into LDS ----
        for (int i = tid; i < (B_ * H_) / 8; i += 256) {          // 4096 x 16B chunks
            const unsigned short* gp = hs_rd + (size_t)i * 8;
            unsigned ldsoff = lds_base + (unsigned)i * 16u;
            asm volatile("global_load_async_to_lds_b128 %0, %1, off"
                         :: "v"(ldsoff), "v"(gp) : "memory");
        }
        asm volatile("s_wait_asynccnt 0x0" ::: "memory");
        __syncthreads();

        // ---- phase 2: C-init from xproj, then WMMA GEMM (A from LDS, B from L2) ----
        v8f acc[4][2];
        for (int rt = 0; rt < 4; ++rt)
            for (int ct = 0; ct < 2; ++ct)
                for (int r = 0; r < 8; ++r) {
                    int b = rt * 16 + r + 8 * hi;
                    acc[rt][ct][r] = xp[((size_t)b * S_ + s) * G4H + n[ct]];
                }

        for (int k0 = 0; k0 < H_; k0 += 32) {
            v16bf bf[2], af[4];
            for (int ct = 0; ct < 2; ++ct) bf[ct] = ld_bfrag(Brow[ct] + k0 + khB);
            for (int rt = 0; rt < 4; ++rt)
                af[rt] = ld_afrag(h_lds + (rt * 16 + mrow) * H_ + k0 + khA);
            for (int rt = 0; rt < 4; ++rt)
                for (int ct = 0; ct < 2; ++ct)
                    acc[rt][ct] = WMMA_BF16(af[rt], bf[ct], acc[rt][ct]);
        }
        __syncthreads();                            // everyone done reading h_lds

        // ---- phase 3: exchange g through LDS (same buffer, now f32 [64][256]) ----
        for (int rt = 0; rt < 4; ++rt)
            for (int ct = 0; ct < 2; ++ct)
                for (int r = 0; r < 8; ++r) {
                    int b = rt * 16 + r + 8 * hi;
                    g_lds[b * 256 + lc[ct]] = acc[rt][ct][r];
                }
        __syncthreads();

        // ---- phase 4: gate nonlinearities, state update ----
        for (int it = 0; it < 16; ++it) {           // 4096 = 64 rows x 64 cols
            int idx = tid + it * 256;
            int b = idx >> 6, j = idx & 63;
            int col = cb * 64 + j;
            float gi = g_lds[b * 256 +       j] + bhh[          col];
            float gf = g_lds[b * 256 +  64 + j] + bhh[    H_ + col];
            float gg = g_lds[b * 256 + 128 + j] + bhh[2 * H_ + col];
            float go = g_lds[b * 256 + 192 + j] + bhh[3 * H_ + col];
            float cp = cs[b * H_ + col];
            float cn = sigm(gf) * cp + sigm(gi) * tanhf(gg);
            float hn = sigm(go) * tanhf(cn);
            cs[b * H_ + col]    = cn;
            hs_wr[b * H_ + col] = f2bf(hn);
            hseq[((size_t)b * S_ + s) * H_ + col] = hn;
        }

        dir_sync(cnt, (unsigned)t + 1u);            // h(t) visible before step t+1 reads it
    }
}

// ---------- attention pooling ----------
__global__ void k_att(const float* __restrict__ hf, const float* __restrict__ hb,
                      const float* __restrict__ myw, float* __restrict__ pooled) {
    const int b = blockIdx.x, tid = threadIdx.x;
    __shared__ float pr[S_];
    __shared__ float red[S_];

    const float* hfr = hf + ((size_t)b * S_ + tid) * H_;
    const float* hbr = hb + ((size_t)b * S_ + tid) * H_;
    float d = 0.0f;
    for (int k = 0; k < H_; ++k) d += hfr[k] * myw[k];
    for (int k = 0; k < H_; ++k) d += hbr[k] * myw[H_ + k];
    float p = __expf(tanhf(d));
    pr[tid] = p; red[tid] = p;
    __syncthreads();
    for (int st = 128; st > 0; st >>= 1) {
        if (tid < st) red[tid] += red[tid + st];
        __syncthreads();
    }
    float denom = red[0];
    pr[tid] = p / denom;
    __syncthreads();

    for (int d0 = tid; d0 < 2 * H_; d0 += 256) {
        const float* src = (d0 < H_) ? (hf + (size_t)b * S_ * H_ + d0)
                                     : (hb + (size_t)b * S_ * H_ + (d0 - H_));
        float acc = 0.0f;
        for (int s2 = 0; s2 < S_; ++s2) acc += pr[s2] * src[(size_t)s2 * H_];
        pooled[b * (2 * H_) + d0] = acc;
    }
}

// ---------- head ----------
__global__ void k_lin1(const float* __restrict__ pooled, const float* __restrict__ w,
                       const float* __restrict__ bias, float* __restrict__ h1) {
    int i = blockIdx.x * 256 + threadIdx.x;        // 64*512
    int b = i >> 9, h = i & 511;
    const float* pr = pooled + b * (2 * H_);
    const float* wr = w + (size_t)h * (2 * H_);
    float acc = bias[h];
    for (int k = 0; k < 2 * H_; ++k) acc += pr[k] * wr[k];
    h1[i] = acc;
}

__global__ void k_lin2(const float* __restrict__ h1, const float* __restrict__ w,
                       const float* __restrict__ bias, float* __restrict__ out) {
    int i = blockIdx.x * 256 + threadIdx.x;        // 64*20
    if (i >= B_ * C_) return;
    int b = i / C_, c = i % C_;
    const float* hr = h1 + b * H_;
    const float* wr = w + (size_t)c * H_;
    float acc = bias[c];
    for (int k = 0; k < H_; ++k) acc += hr[k] * wr[k];
    out[i] = acc;
}

// ---------- launch ----------
extern "C" void kernel_launch(void* const* d_in, const int* in_sizes, int n_in,
                              void* d_out, int out_size, void* d_ws, size_t ws_size,
                              hipStream_t stream) {
    const int*   x      = (const int*)  d_in[0];
    const float* embed  = (const float*)d_in[1];
    const float* w_ih_f = (const float*)d_in[2];
    const float* w_hh_f = (const float*)d_in[3];
    const float* b_ih_f = (const float*)d_in[4];
    const float* b_hh_f = (const float*)d_in[5];
    const float* w_ih_b = (const float*)d_in[6];
    const float* w_hh_b = (const float*)d_in[7];
    const float* b_ih_b = (const float*)d_in[8];
    const float* b_hh_b = (const float*)d_in[9];
    const float* myw    = (const float*)d_in[10];
    const float* lin1_w = (const float*)d_in[11];
    const float* lin1_b = (const float*)d_in[12];
    const float* lin2_w = (const float*)d_in[13];
    const float* lin2_b = (const float*)d_in[14];

    char* ws = (char*)d_ws;
    size_t off = 0;
    auto alloc = [&](size_t bytes) -> void* {
        void* p = ws + off;
        off = (off + bytes + 255) & ~(size_t)255;
        return p;
    };
    unsigned short* embB  = (unsigned short*)alloc((size_t)B_ * S_ * E_ * 2);
    unsigned short* wihfB = (unsigned short*)alloc((size_t)G4H * E_ * 2);
    unsigned short* wihbB = (unsigned short*)alloc((size_t)G4H * E_ * 2);
    unsigned short* whhfB = (unsigned short*)alloc((size_t)G4H * H_ * 2);
    unsigned short* whhbB = (unsigned short*)alloc((size_t)G4H * H_ * 2);
    float* xpf            = (float*)alloc((size_t)B_ * S_ * G4H * 4);
    float* xpb            = (float*)alloc((size_t)B_ * S_ * G4H * 4);
    unsigned short* hstat = (unsigned short*)alloc((size_t)4 * B_ * H_ * 2); // 2 dirs x 2 bufs
    float* cstat          = (float*)alloc((size_t)2 * B_ * H_ * 4);
    float* hseqf          = (float*)alloc((size_t)B_ * S_ * H_ * 4);
    float* hseqb          = (float*)alloc((size_t)B_ * S_ * H_ * 4);
    float* pooled         = (float*)alloc((size_t)B_ * 2 * H_ * 4);
    float* h1             = (float*)alloc((size_t)B_ * H_ * 4);
    unsigned int* sync    = (unsigned int*)alloc(256);

    const int NW = G4H * E_;                       // 1048576 elems per weight matrix
    k_f2bf<<<NW / 256, 256, 0, stream>>>(w_ih_f, wihfB, NW);
    k_f2bf<<<NW / 256, 256, 0, stream>>>(w_ih_b, wihbB, NW);
    k_f2bf<<<NW / 256, 256, 0, stream>>>(w_hh_f, whhfB, NW);
    k_f2bf<<<NW / 256, 256, 0, stream>>>(w_hh_b, whhbB, NW);
    k_embed<<<(B_ * S_ * E_) / 256, 256, 0, stream>>>(x, embed, embB);
    k_init<<<(4 * B_ * H_ + 255) / 256, 256, 0, stream>>>(hstat, cstat, sync);

    k_xproj<<<dim3(G4H / 256, (B_ * S_) / 64, 2), 256, 0, stream>>>(
        embB, wihfB, wihbB, b_ih_f, b_ih_b, xpf, xpb);

    k_recurrent<<<dim3(H_ / 64, 2), 256, 0, stream>>>(
        xpf, xpb, whhfB, whhbB, b_hh_f, b_hh_b, hstat, cstat, hseqf, hseqb, sync);

    k_att<<<B_, 256, 0, stream>>>(hseqf, hseqb, myw, pooled);
    k_lin1<<<(B_ * H_) / 256, 256, 0, stream>>>(pooled, lin1_w, lin1_b, h1);
    k_lin2<<<(B_ * C_ + 255) / 256, 256, 0, stream>>>(h1, lin2_w, lin2_b, (float*)d_out);
}
